// Scaled_Dot_Product_Cross_Attention_53077205844766
// MI455X (gfx1250) — matrix operand order repaired and checked by reading, not compiled, not run on hardware
//
#include <hip/hip_runtime.h>

// ---------------- problem sizes ----------------
constexpr int Bb = 4;
constexpr int Nn = 4096;
constexpr int Cc = 1024;
constexpr int Dd = 1024;

// ---------------- tiling ----------------
constexpr int TM = 128;   // macro tile rows
constexpr int TN = 128;   // macro tile cols
constexpr int TK = 64;    // K step per LDS stage
constexpr int LDT = 72;   // padded LDS row stride (bf16 elems): 144B, 16B-aligned

typedef __attribute__((ext_vector_type(16))) __bf16 bf16x16;
typedef __attribute__((ext_vector_type(8)))  float  f32x8;
typedef unsigned int u32x4 __attribute__((ext_vector_type(4)));
typedef int i32x8 __attribute__((ext_vector_type(8)));
typedef int i32x4 __attribute__((ext_vector_type(4)));

union FragU { bf16x16 v; uint4 q[2]; };
union Pack8 { uint4 q; __bf16 h[8]; };

__device__ __forceinline__ __bf16 to_bf16(float f) { return (__bf16)f; }

// Async global->LDS 16-byte copy (GLOBAL_LOAD_ASYNC_TO_LDS_B128, ASYNCcnt).
// LDS destination address = low 32 bits of the generic shared pointer.
__device__ __forceinline__ void async_ld16(void* ldsPtr, const void* gPtr) {
  const unsigned lds = (unsigned)(uintptr_t)ldsPtr;
  const unsigned long long ga = (unsigned long long)(uintptr_t)gPtr;
  asm volatile("global_load_async_to_lds_b128 %0, %1, off"
               :: "v"(lds), "v"(ga)
               : "memory");
}

__device__ __forceinline__ void wait_async0() {
#if __has_builtin(__builtin_amdgcn_s_wait_asynccnt)
  __builtin_amdgcn_s_wait_asynccnt(0);
#else
  asm volatile("s_wait_asynccnt 0x0" ::: "memory");
#endif
}

// ------------------------------------------------------------------
// Tensor Data Mover: one per-wave DMA of a 128(row) x 64(col) bf16 tile
// from a row-major tensor (row stride = `stride` elems) into LDS.
// D# pad fields make HW insert 8 bf16 of padding after each 64-bf16 row:
//   pad_interval code 4 = 32 DWORDs (=128B row), pad_amount code 3 = 4 DWORDs
// -> LDS row stride 72 bf16 == LDT.
// ------------------------------------------------------------------
__device__ __forceinline__ void tdm_load_tile_bf16(void* ldsDst, const void* gSrc,
                                                   int stride) {
#if __has_builtin(__builtin_amdgcn_tensor_load_to_lds)
  const unsigned lds = (unsigned)(uintptr_t)ldsDst;
  const unsigned long long ga = (unsigned long long)(uintptr_t)gSrc;
  u32x4 g0;
  g0.x = 1u;                                                  // count=1 (user D#)
  g0.y = lds;                                                 // lds_addr
  g0.z = (unsigned)(ga & 0xFFFFFFFFull);                      // global_addr[31:0]
  g0.w = (unsigned)((ga >> 32) & 0x1FFFFFFull) | (2u << 30);  // addr[56:32] | type=2
  i32x8 g1;
  g1[0] = (1 << 16) | (1 << 20) | (4 << 22) | (3 << 25);  // data_size=2B, pad_en,
                                                          // interval=32dw, pad=4dw
  g1[1] = (stride & 0xFFFF) << 16;  // tensor_dim0[15:0] in bits[31:16]
  g1[2] = (Nn & 0xFFFF) << 16;      // dim0 hi=0 | tensor_dim1[15:0]
  g1[3] = (TK << 16);               // dim1 hi=0 | tile_dim0 = 64
  g1[4] = TM;                       // tile_dim1 = 128 | tile_dim2 = 0
  g1[5] = stride;                   // tensor_dim0_stride[31:0]
  g1[6] = 0;                        // stride0 hi | tensor_dim1_stride lo
  g1[7] = 0;
  const i32x4 z4 = {0, 0, 0, 0};
  const i32x8 z8 = {0, 0, 0, 0, 0, 0, 0, 0};
  // 6-arg form (clang-23 / therock-10.0 headers)
  __builtin_amdgcn_tensor_load_to_lds(g0, g1, z4, z4, z8, 0);
#else
  (void)ldsDst; (void)gSrc; (void)stride;
#endif
}

__device__ __forceinline__ void wait_tensor0() {
#if __has_builtin(__builtin_amdgcn_s_wait_tensorcnt)
  __builtin_amdgcn_s_wait_tensorcnt(0);
#else
  asm volatile("s_wait_tensorcnt 0x0" ::: "memory");
#endif
}

// A-fragment (16x32 bf16). ISA layout: lane L -> M=L%16, half=L/16.
// elems 0..7  : K = kk + 8*half + (0..7)      (16 contiguous bytes)
// elems 8..15 : K = kk + 16 + 8*half + (0..7) (16 contiguous bytes)
__device__ __forceinline__ bf16x16 load_frag_a(const __bf16* lds, int row0, int kk) {
  const int lane = threadIdx.x & 31;
  const int m = lane & 15, half = lane >> 4;
  const __bf16* p = lds + (row0 + m) * LDT + kk + half * 8;
  FragU u;
  u.q[0] = *(const uint4*)(p);
  u.q[1] = *(const uint4*)(p + 16);
  return u.v;
}

// B-fragment (32x16 bf16), LDS tile stored transposed as [N][K]:
// lane L -> N=L%16, half=L/16; elems 0..7: K = kk + 8*half + e, elems 8..15: K = kk + 16 + 8*half + (e-8)
__device__ __forceinline__ bf16x16 load_frag_b(const __bf16* lds, int col0, int kk) {
  const int lane = threadIdx.x & 31;
  const int n = lane & 15, half = lane >> 4;
  const __bf16* p = lds + (col0 + n) * LDT + kk + half * 8;
  FragU u;
  u.q[0] = *(const uint4*)(p);
  u.q[1] = *(const uint4*)(p + 16);
  return u.v;
}

// One LDS stage: 2 K-steps of 32, wave computes a 32x64 tile (2x4 of 16x16)
__device__ __forceinline__ void mma_stage(const __bf16* ldsA, const __bf16* ldsB,
                                          int wrow, int wcol, f32x8 (&acc)[2][4]) {
#pragma unroll
  for (int kk = 0; kk < TK; kk += 32) {
    bf16x16 a[2], b[4];
    a[0] = load_frag_a(ldsA, wrow + 0, kk);
    a[1] = load_frag_a(ldsA, wrow + 16, kk);
#pragma unroll
    for (int ni = 0; ni < 4; ++ni) b[ni] = load_frag_b(ldsB, wcol + 16 * ni, kk);
#pragma unroll
    for (int mi = 0; mi < 2; ++mi)
#pragma unroll
      for (int ni = 0; ni < 4; ++ni)
        acc[mi][ni] = __builtin_amdgcn_wmma_f32_16x16x32_bf16(
            false, a[mi], false, b[ni], (short)0, acc[mi][ni], false, false);
  }
}

__device__ __forceinline__ void zero_acc(f32x8 (&acc)[2][4]) {
  f32x8 z = {0.f, 0.f, 0.f, 0.f, 0.f, 0.f, 0.f, 0.f};
#pragma unroll
  for (int mi = 0; mi < 2; ++mi)
#pragma unroll
    for (int ni = 0; ni < 4; ++ni) acc[mi][ni] = z;
}

// ------------------------------------------------------------------
// Kernel 1: projections. z selects {0: Q=X*Wq, 1: K=Y*Wk, 2: Vx=X*Wv, 3: Vy=Y*Wv}
// Treats X as (B*N, C) row-major; GEMM (B*N,C)x(C,D).
// ------------------------------------------------------------------
__global__ __launch_bounds__(256) void proj_kernel(
    const float* __restrict__ x, const float* __restrict__ y,
    const float* __restrict__ Wq, const float* __restrict__ Wk,
    const float* __restrict__ Wv,
    __bf16* __restrict__ qbf, __bf16* __restrict__ kbf,
    __bf16* __restrict__ vxbf, __bf16* __restrict__ vybf,
    float* __restrict__ vxf, float* __restrict__ vyf) {
  __shared__ __bf16 ldsA[TM * LDT];
  __shared__ __bf16 ldsB[TN * LDT];

  const int which = blockIdx.z;
  const float* src = (which == 1 || which == 3) ? y : x;
  const float* W = (which == 0) ? Wq : (which == 1) ? Wk : Wv;
  __bf16* obf = (which == 0) ? qbf : (which == 1) ? kbf : (which == 2) ? vxbf : vybf;
  float* of32 = (which == 2) ? vxf : (which == 3) ? vyf : nullptr;

  const size_t n0 = (size_t)blockIdx.x * TN;
  const size_t m0 = (size_t)blockIdx.y * TM;
  const int wave = threadIdx.x >> 5;
  const int wrow = (wave & 3) * 32;
  const int wcol = (wave >> 2) * 64;

  f32x8 acc[2][4];
  zero_acc(acc);

  for (int k0 = 0; k0 < Cc; k0 += TK) {
    if (k0 + TK < Cc) {  // pull next K tile toward L2 (global_prefetch_b8)
      __builtin_prefetch(&src[(m0 + (threadIdx.x >> 1)) * Cc + k0 + TK], 0, 1);
      __builtin_prefetch(&W[(size_t)(k0 + TK + (threadIdx.x >> 2)) * Dd + n0], 0, 1);
    }
    // A tile: 128 x 64 fp32 -> bf16, layout [m][k]
#pragma unroll
    for (int i = 0; i < 8; ++i) {
      const int flat = i * 256 + threadIdx.x;
      const int row = flat >> 4, c4 = flat & 15;
      const float4 f = *(const float4*)&src[(m0 + row) * Cc + k0 + c4 * 4];
      __bf16* d = &ldsA[row * LDT + c4 * 4];
      d[0] = to_bf16(f.x); d[1] = to_bf16(f.y); d[2] = to_bf16(f.z); d[3] = to_bf16(f.w);
    }
    // B tile: W[k][n] 64 x 128 -> transposed LDS [n][k]
#pragma unroll
    for (int i = 0; i < 8; ++i) {
      const int flat = i * 256 + threadIdx.x;
      const int row = flat >> 5, c4 = flat & 31;
      const float4 f = *(const float4*)&W[(size_t)(k0 + row) * Dd + n0 + c4 * 4];
      ldsB[(c4 * 4 + 0) * LDT + row] = to_bf16(f.x);
      ldsB[(c4 * 4 + 1) * LDT + row] = to_bf16(f.y);
      ldsB[(c4 * 4 + 2) * LDT + row] = to_bf16(f.z);
      ldsB[(c4 * 4 + 3) * LDT + row] = to_bf16(f.w);
    }
    __syncthreads();
    mma_stage(ldsA, ldsB, wrow, wcol, acc);
    __syncthreads();
  }

  const int lane = threadIdx.x & 31;
  const int n = lane & 15, half = lane >> 4;
#pragma unroll
  for (int mi = 0; mi < 2; ++mi)
#pragma unroll
    for (int ni = 0; ni < 4; ++ni) {
      const size_t row = m0 + wrow + mi * 16 + half * 8;
      const size_t col = n0 + wcol + ni * 16 + n;
#pragma unroll
      for (int r = 0; r < 8; ++r) {
        const float vv = acc[mi][ni][r];
        obf[(row + r) * Dd + col] = to_bf16(vv);
        if (of32) of32[(row + r) * Dd + col] = vv;
      }
    }
}

// ------------------------------------------------------------------
// Kernel 2: energy[b] = Q_b (N x D) * K_b^T  -> fp32 (N x N)
// A (Q) tile: TDM tensor_load_to_lds issued by wave 0 (TENSORcnt).
// B (K) tile: per-lane GLOBAL_LOAD_ASYNC_TO_LDS_B128 (ASYNCcnt).
// ------------------------------------------------------------------
__global__ __launch_bounds__(256) void energy_kernel(const __bf16* __restrict__ qbf,
                                                     const __bf16* __restrict__ kbf,
                                                     float* __restrict__ E) {
  __shared__ __bf16 ldsA[TM * LDT];
  __shared__ __bf16 ldsB[TN * LDT];

  const int b = blockIdx.z;
  const size_t n0 = (size_t)blockIdx.y * TM;  // rows of E (Q rows)
  const size_t m0 = (size_t)blockIdx.x * TN;  // cols of E (K rows)
  const __bf16* Q = qbf + (size_t)b * Nn * Dd;
  const __bf16* Kp = kbf + (size_t)b * Nn * Dd;
  float* Eb = E + (size_t)b * Nn * Nn;

  const int wave = threadIdx.x >> 5;
  const int wrow = (wave & 3) * 32;
  const int wcol = (wave >> 2) * 64;

  f32x8 acc[2][4];
  zero_acc(acc);

  for (int k0 = 0; k0 < Dd; k0 += TK) {
    if (k0 + TK < Dd) {
      __builtin_prefetch(&Q[(n0 + (threadIdx.x >> 1)) * Dd + k0 + TK], 0, 1);
      __builtin_prefetch(&Kp[(m0 + (threadIdx.x >> 1)) * Dd + k0 + TK], 0, 1);
    }
#if __has_builtin(__builtin_amdgcn_tensor_load_to_lds)
    // A tile via Tensor Data Mover: one wave-level DMA for the whole 128x64 tile
    if (wave == 0) tdm_load_tile_bf16(ldsA, &Q[n0 * Dd + k0], Dd);
#else
#pragma unroll
    for (int i = 0; i < 4; ++i) {
      const int flat = i * 256 + threadIdx.x;
      const int row = flat >> 3, c8 = flat & 7;
      async_ld16(&ldsA[row * LDT + c8 * 8], &Q[(n0 + row) * Dd + k0 + c8 * 8]);
    }
#endif
    // B tile via per-lane async copies: 4x16B per lane
#pragma unroll
    for (int i = 0; i < 4; ++i) {
      const int flat = i * 256 + threadIdx.x;
      const int row = flat >> 3, c8 = flat & 7;
      async_ld16(&ldsB[row * LDT + c8 * 8], &Kp[(m0 + row) * Dd + k0 + c8 * 8]);
    }
    wait_async0();  // this wave's async LDS writes have landed
#if __has_builtin(__builtin_amdgcn_tensor_load_to_lds)
    if (wave == 0) wait_tensor0();  // TDM transfer complete
#endif
    __syncthreads();  // publish LDS across waves
    mma_stage(ldsA, ldsB, wrow, wcol, acc);
    __syncthreads();
  }

  const int lane = threadIdx.x & 31;
  const int n = lane & 15, half = lane >> 4;
#pragma unroll
  for (int mi = 0; mi < 2; ++mi)
#pragma unroll
    for (int ni = 0; ni < 4; ++ni) {
      const size_t row = n0 + wrow + mi * 16 + half * 8;
      const size_t col = m0 + wcol + ni * 16 + n;
#pragma unroll
      for (int r = 0; r < 8; ++r) Eb[(row + r) * Nn + col] = acc[mi][ni][r];
    }
}

// ------------------------------------------------------------------
// Kernel 3a: per-row max and 1/sum(exp) of E  (softmax over axis -1)
// ------------------------------------------------------------------
__global__ __launch_bounds__(256) void row_stats_kernel(const float* __restrict__ E,
                                                        float* __restrict__ rmax,
                                                        float* __restrict__ rinv) {
  __shared__ float red[256];
  const int tid = threadIdx.x;
  const int n = blockIdx.x, b = blockIdx.y;
  const float* row = E + ((size_t)b * Nn + n) * Nn;

  float m = -3.4e38f;
  for (int i = tid; i < Nn; i += 256) m = fmaxf(m, row[i]);
  red[tid] = m;
  __syncthreads();
  for (int s = 128; s > 0; s >>= 1) {
    if (tid < s) red[tid] = fmaxf(red[tid], red[tid + s]);
    __syncthreads();
  }
  m = red[0];
  __syncthreads();

  float sum = 0.f;
  for (int i = tid; i < Nn; i += 256) sum += __expf(row[i] - m);
  red[tid] = sum;
  __syncthreads();
  for (int s = 128; s > 0; s >>= 1) {
    if (tid < s) red[tid] += red[tid + s];
    __syncthreads();
  }
  if (tid == 0) {
    rmax[(size_t)b * Nn + n] = m;
    rinv[(size_t)b * Nn + n] = 1.0f / red[0];
  }
}

// ------------------------------------------------------------------
// Kernel 3b: per-column max and 1/sum(exp) of E (softmax of E^T rows)
// thread t owns column blockIdx.x*256+t  -> fully coalesced loads
// ------------------------------------------------------------------
__global__ __launch_bounds__(256) void col_stats_kernel(const float* __restrict__ E,
                                                        float* __restrict__ cmax,
                                                        float* __restrict__ cinv) {
  const int b = blockIdx.y;
  const int col = blockIdx.x * 256 + threadIdx.x;
  const float* base = E + (size_t)b * Nn * Nn + col;

  float m = -3.4e38f;
  for (int r = 0; r < Nn; ++r) m = fmaxf(m, base[(size_t)r * Nn]);
  float s = 0.f;
  for (int r = 0; r < Nn; ++r) s += __expf(base[(size_t)r * Nn] - m);
  cmax[(size_t)b * Nn + col] = m;
  cinv[(size_t)b * Nn + col] = 1.0f / s;
}

// ------------------------------------------------------------------
// Kernel 4: out = sa * (softmax(E or E^T) @ V) + sv * Vf32
//   colMode=0 (out_x): A[n][m]=exp(E[n][m]-rmax[n])*rinv[n]; sa=b^2, sv=1+b
//   colMode=1 (out_y): A[n][m]=exp(E[m][n]-cmax[n])*cinv[n]; sa=b,   sv=1
// ------------------------------------------------------------------
__global__ __launch_bounds__(256) void attn_out_kernel(
    const float* __restrict__ E, const __bf16* __restrict__ vbf,
    const float* __restrict__ vf32, const float* __restrict__ smax,
    const float* __restrict__ sinv, const float* __restrict__ beta,
    float* __restrict__ out, int colMode) {
  __shared__ __bf16 ldsA[TM * LDT];
  __shared__ __bf16 ldsB[TN * LDT];

  const int b = blockIdx.z;
  const size_t n0 = (size_t)blockIdx.y * TM;  // out rows
  const size_t d0 = (size_t)blockIdx.x * TN;  // out cols
  const float* Eb = E + (size_t)b * Nn * Nn;
  const __bf16* Vb = vbf + (size_t)b * Nn * Dd;
  const float* Vf = vf32 + (size_t)b * Nn * Dd;
  const float* mx = smax + (size_t)b * Nn;
  const float* iv = sinv + (size_t)b * Nn;

  const int wave = threadIdx.x >> 5;
  const int wrow = (wave & 3) * 32;
  const int wcol = (wave >> 2) * 64;

  f32x8 acc[2][4];
  zero_acc(acc);

  for (int k0 = 0; k0 < Nn; k0 += TK) {
    if (!colMode) {
      // A tile 128(n) x 64(m): read E rows, softmax, -> bf16 [n][m]
#pragma unroll
      for (int i = 0; i < 8; ++i) {
        const int flat = i * 256 + threadIdx.x;
        const int row = flat >> 4, c4 = flat & 15;
        const float4 e = *(const float4*)&Eb[(n0 + row) * Nn + k0 + c4 * 4];
        const float M = mx[n0 + row], I = iv[n0 + row];
        __bf16* d = &ldsA[row * LDT + c4 * 4];
        d[0] = to_bf16(__expf(e.x - M) * I);
        d[1] = to_bf16(__expf(e.y - M) * I);
        d[2] = to_bf16(__expf(e.z - M) * I);
        d[3] = to_bf16(__expf(e.w - M) * I);
      }
    } else {
      // A tile 128(n) x 64(m): read E[m][n] rows (coalesced), transpose into [n][m]
#pragma unroll
      for (int i = 0; i < 8; ++i) {
        const int flat = i * 256 + threadIdx.x;
        const int row = flat >> 5, c4 = flat & 31;  // row: m-local (64), c4: n/4 (32)
        const float4 e = *(const float4*)&Eb[(size_t)(k0 + row) * Nn + n0 + c4 * 4];
        const float ev[4] = {e.x, e.y, e.z, e.w};
#pragma unroll
        for (int j = 0; j < 4; ++j) {
          const size_t ncol = n0 + c4 * 4 + j;
          ldsA[(c4 * 4 + j) * LDT + row] = to_bf16(__expf(ev[j] - mx[ncol]) * iv[ncol]);
        }
      }
    }
    // B tile: V[k0..+63][d0..+127] bf16 -> transposed LDS [d][m]
#pragma unroll
    for (int i = 0; i < 4; ++i) {
      const int flat = i * 256 + threadIdx.x;
      const int row = flat >> 4, c8 = flat & 15;
      Pack8 p;
      p.q = *(const uint4*)&Vb[(size_t)(k0 + row) * Dd + d0 + c8 * 8];
#pragma unroll
      for (int j = 0; j < 8; ++j) ldsB[(c8 * 8 + j) * LDT + row] = p.h[j];
    }
    __syncthreads();
    mma_stage(ldsA, ldsB, wrow, wcol, acc);
    __syncthreads();
  }

  const float bet = beta[0];
  const float sa = colMode ? bet : bet * bet;
  const float sv = colMode ? 1.0f : 1.0f + bet;
  const int lane = threadIdx.x & 31;
  const int n = lane & 15, half = lane >> 4;
#pragma unroll
  for (int mi = 0; mi < 2; ++mi)
#pragma unroll
    for (int ni = 0; ni < 4; ++ni) {
      const size_t row = n0 + wrow + mi * 16 + half * 8;
      const size_t col = d0 + wcol + ni * 16 + n;
#pragma unroll
      for (int r = 0; r < 8; ++r) {
        const size_t idx = (row + r) * Dd + col;
        out[idx] = sa * acc[mi][ni][r] + sv * Vf[idx];
      }
    }
}

// ------------------------------------------------------------------
extern "C" void kernel_launch(void* const* d_in, const int* in_sizes, int n_in,
                              void* d_out, int out_size, void* d_ws, size_t ws_size,
                              hipStream_t stream) {
  (void)in_sizes; (void)n_in; (void)out_size; (void)ws_size;
  const float* x = (const float*)d_in[0];
  const float* y = (const float*)d_in[1];
  const float* Wq = (const float*)d_in[2];
  const float* Wk = (const float*)d_in[3];
  const float* Wv = (const float*)d_in[4];
  const float* beta = (const float*)d_in[5];

  const size_t P = (size_t)Bb * Nn * Dd;  // 16,777,216 elements

  // workspace layout
  __bf16* qbf = (__bf16*)d_ws;
  __bf16* kbf = qbf + P;
  __bf16* vxbf = kbf + P;
  __bf16* vybf = vxbf + P;
  float* vxf = (float*)(vybf + P);
  float* vyf = vxf + P;
  float* E = vyf + P;  // Bb*Nn*Nn fp32 (256 MiB; 64 MiB per batch -> L2 resident)
  float* rmax = E + (size_t)Bb * Nn * Nn;
  float* rinv = rmax + (size_t)Bb * Nn;
  float* cmax = rinv + (size_t)Bb * Nn;
  float* cinv = cmax + (size_t)Bb * Nn;

  float* outx = (float*)d_out;
  float* outy = outx + P;

  dim3 blk(256);
  proj_kernel<<<dim3(Dd / TN, (Bb * Nn) / TM, 4), blk, 0, stream>>>(
      x, y, Wq, Wk, Wv, qbf, kbf, vxbf, vybf, vxf, vyf);
  energy_kernel<<<dim3(Nn / TN, Nn / TM, Bb), blk, 0, stream>>>(qbf, kbf, E);
  row_stats_kernel<<<dim3(Nn, Bb), blk, 0, stream>>>(E, rmax, rinv);
  col_stats_kernel<<<dim3(Nn / 256, Bb), blk, 0, stream>>>(E, cmax, cinv);
  attn_out_kernel<<<dim3(Dd / TN, Nn / TM, Bb), blk, 0, stream>>>(
      E, vxbf, vxf, rmax, rinv, beta, outx, 0);
  attn_out_kernel<<<dim3(Dd / TN, Nn / TM, Bb), blk, 0, stream>>>(
      E, vybf, vyf, cmax, cinv, beta, outy, 1);
}
